// GraphNN_14989435863156
// MI455X (gfx1250) — matrix operand order repaired and checked by reading, not compiled, not run on hardware
//
#include <hip/hip_runtime.h>
#include <hip/hip_bf16.h>
#include <math.h>

// ---------------------------------------------------------------------------
// Problem constants (fixed by the reference)
// ---------------------------------------------------------------------------
#define NN     2048      // number of instructions
#define TT     16        // tokens per instruction
#define EE     512       // token embedding dim
#define HID    256       // hidden size
#define G4     1024      // 4*HID (gate width)
#define PP     8         // max parents
#define LEVELS 16
#define KLVL   (NN / LEVELS)   // 128 instructions per level

typedef _Float16 half8  __attribute__((ext_vector_type(8)));
typedef _Float16 v16h   __attribute__((ext_vector_type(16)));
typedef float    v8f    __attribute__((ext_vector_type(8)));

// ---------------------------------------------------------------------------
// Weight/bias permutation: original gate rows are g*256+j (g: 0=i,1=f,2=g,3=o)
// permuted row n' = j*4+g so that 64 consecutive output columns hold 16
// complete (i,f,g,o) quadruples -> cell update is block-local.
// ---------------------------------------------------------------------------
__global__ void k_permute_w(const float* __restrict__ src, _Float16* __restrict__ dst, int K) {
    int i = blockIdx.x * blockDim.x + threadIdx.x;
    if (i >= G4 * K) return;
    int np = i / K, k = i - np * K;
    int j = np >> 2, g = np & 3;
    dst[i] = (_Float16)src[(size_t)(g * HID + j) * K + k];
}

__global__ void k_permute_bias(const float* __restrict__ b1, const float* __restrict__ b2,
                               float* __restrict__ dst) {
    int i = blockIdx.x * blockDim.x + threadIdx.x;
    if (i >= G4) return;
    int j = i >> 2, g = i & 3;
    dst[i] = b1[g * HID + j] + b2[g * HID + j];
}

__global__ void k_zero_f32(float* __restrict__ dst, int n) {
    int i = blockIdx.x * blockDim.x + threadIdx.x;
    if (i < n) dst[i] = 0.0f;
}

__global__ void k_zero_i32(int* __restrict__ dst, int n) {
    int i = blockIdx.x * blockDim.x + threadIdx.x;
    if (i < n) dst[i] = 0;
}

// ---------------------------------------------------------------------------
// WMMA mainloop: acc += A[M,K] @ Wp[1024,K]^T  (per-block 64x64 tile)
// 256 threads = 8 wave32; wave (wm,wn) owns one 16-row x two 16-col tiles.
// ---------------------------------------------------------------------------
#define LDSS 40   // LDS row stride in halves: 80B -> 16B-aligned ds_load_b128

__device__ __forceinline__ void gemm_pass(
    const float* __restrict__ A, int lda, const _Float16* __restrict__ W, int K,
    int rb, int n0, int tid, _Float16* As, _Float16* Bs, v8f& acc0, v8f& acc1)
{
    const int lane = tid & 31;
    const int wid  = tid >> 5;
    const int wm   = wid & 3;        // 16-row sub-tile of M
    const int wn   = wid >> 2;       // 32-col sub-tile of N
    const int r    = tid >> 2;       // 0..63 : row staged by this thread
    const int kk   = (tid & 3) * 8;  // 0,8,16,24 : k offset staged

    const int arow = wm * 16 + (lane & 15);
    const int kbA  = (lane >> 4) * 8;   // A frag: e0..7=K[kbA..+7], e8..15=K[kbA+16..+23]
    const int bcol = lane & 15;
    const int kbB  = (lane >> 4) * 8;   // B frag: e0..7=K[kbB..+7], e8..15=K[kbB+16..+23]

    for (int k0 = 0; k0 < K; k0 += 32) {
        __syncthreads();
        // ---- stage A tile (f32 -> f16) ----
        const float* ap = A + (size_t)(rb + r) * lda + k0 + kk;
        float4 f0 = *(const float4*)ap;
        float4 f1 = *(const float4*)(ap + 4);
        _Float16* as = &As[r * LDSS + kk];
        as[0] = (_Float16)f0.x; as[1] = (_Float16)f0.y;
        as[2] = (_Float16)f0.z; as[3] = (_Float16)f0.w;
        as[4] = (_Float16)f1.x; as[5] = (_Float16)f1.y;
        as[6] = (_Float16)f1.z; as[7] = (_Float16)f1.w;
        // ---- stage W tile (f16 row-major [n'][k]) ----
        *(half8*)&Bs[r * LDSS + kk] =
            *(const half8*)(W + (size_t)(n0 + r) * K + k0 + kk);
        __syncthreads();

        half8 alo = *(const half8*)&As[arow * LDSS + kbA];
        half8 ahi = *(const half8*)&As[arow * LDSS + kbA + 16];
        v16h a;
#pragma unroll
        for (int i = 0; i < 8; ++i) { a[i] = alo[i]; a[i + 8] = ahi[i]; }

        const int br0 = wn * 32 + bcol;
        half8 b0lo = *(const half8*)&Bs[br0 * LDSS + kbB];
        half8 b0hi = *(const half8*)&Bs[br0 * LDSS + kbB + 16];
        v16h b0;
#pragma unroll
        for (int i = 0; i < 8; ++i) { b0[i] = b0lo[i]; b0[i + 8] = b0hi[i]; }

        const int br1 = br0 + 16;
        half8 b1lo = *(const half8*)&Bs[br1 * LDSS + kbB];
        half8 b1hi = *(const half8*)&Bs[br1 * LDSS + kbB + 16];
        v16h b1;
#pragma unroll
        for (int i = 0; i < 8; ++i) { b1[i] = b1lo[i]; b1[i + 8] = b1hi[i]; }

        acc0 = __builtin_amdgcn_wmma_f32_16x16x32_f16(false, a, false, b0,
                                                      (short)0, acc0, false, false);
        acc1 = __builtin_amdgcn_wmma_f32_16x16x32_f16(false, a, false, b1,
                                                      (short)0, acc1, false, false);
    }
}

__device__ __forceinline__ float sigf(float x) { return 1.0f / (1.0f + __expf(-x)); }

// ---------------------------------------------------------------------------
// Fused GEMM + LSTM cell.
//   pre-act = A1 @ W1p^T + A2 @ W2p^T + bsum   (permuted gate columns)
//   block (x,y): rows x*64..+63, hidden units y*16..+15  (disjoint ownership)
//   c_new = sig(f)*c_in + sig(i)*tanh(g);  h = sig(o)*tanh(c_new)
// h_out must NOT alias A2 (caller ping-pongs h); c_in may alias c_out.
// ---------------------------------------------------------------------------
#define CSTR 68   // f32 LDS row stride for the 64x64 epilogue tile

__global__ __launch_bounds__(256)
void k_gemm_lstm(const float* __restrict__ A1, int lda1,
                 const _Float16* __restrict__ W1, int K1,
                 const float* __restrict__ A2, int lda2,
                 const _Float16* __restrict__ W2, int K2,
                 const float* __restrict__ bsum,
                 const float* __restrict__ c_in,
                 float* __restrict__ h_out,
                 float* __restrict__ c_out)
{
    __shared__ __align__(16) char smem[64 * CSTR * 4];   // 17408 B
    _Float16* As = (_Float16*)smem;                      // 64*LDSS halves = 5120 B
    _Float16* Bs = (_Float16*)(smem + 64 * LDSS * 2);    // 5120 B
    float*    Cs = (float*)smem;                         // reused after mainloop

    const int tid = threadIdx.x;
    const int rb  = blockIdx.x * 64;   // M base
    const int n0  = blockIdx.y * 64;   // permuted-gate column base

    v8f acc0 = {};
    v8f acc1 = {};
    gemm_pass(A1, lda1, W1, K1, rb, n0, tid, As, Bs, acc0, acc1);
    gemm_pass(A2, lda2, W2, K2, rb, n0, tid, As, Bs, acc0, acc1);

    __syncthreads();   // all waves done reading As/Bs before aliasing as Cs

    // ---- spill accumulators (+bias) to LDS ----
    const int lane = tid & 31;
    const int wid  = tid >> 5;
    const int wm   = wid & 3;
    const int wn   = wid >> 2;
    const int mofs = (lane >> 4) * 8;
    const int ncol = lane & 15;
    const float* bp = bsum + n0;
#pragma unroll
    for (int v = 0; v < 8; ++v) {
        const int ml = wm * 16 + v + mofs;
        const int nl = wn * 32 + ncol;
        Cs[ml * CSTR + nl]      = acc0[v] + bp[nl];
        Cs[ml * CSTR + nl + 16] = acc1[v] + bp[nl + 16];
    }
    __syncthreads();

    // ---- LSTM cell update: 64 rows x 16 hidden units, 4 per thread ----
#pragma unroll
    for (int u = 0; u < 4; ++u) {
        const int e   = tid + u * 256;     // 0..1023
        const int row = e >> 4;            // 0..63
        const int jl  = e & 15;            // 0..15
        const float* q = &Cs[row * CSTR + jl * 4];
        float ig = q[0], fg = q[1], gg = q[2], og = q[3];
        const size_t ci = (size_t)(rb + row) * HID + (blockIdx.y * 16 + jl);
        float cn = sigf(fg) * c_in[ci] + sigf(ig) * tanhf(gg);
        c_out[ci] = cn;
        h_out[ci] = sigf(og) * tanhf(cn);
    }
}

// ---------------------------------------------------------------------------
// Parent gather with max-combine (one block per instruction in the level)
// ---------------------------------------------------------------------------
__global__ void k_gather_parents(const int* __restrict__ pidx,
                                 const unsigned char* __restrict__ pval,
                                 const float* __restrict__ H,
                                 const float* __restrict__ C,
                                 float* __restrict__ h0,
                                 float* __restrict__ c0, int base_n)
{
    const int k  = blockIdx.x;          // 0..127
    const int hd = threadIdx.x;         // 0..255
    const int n  = base_n + k;
    float hm = -1e30f, cm = -1e30f;
    int any = 0;
#pragma unroll
    for (int p = 0; p < PP; ++p) {
        int pi = pidx[n * PP + p];
        if (pval[n * PP + p]) {
            any = 1;
            hm = fmaxf(hm, H[(size_t)pi * HID + hd]);
            cm = fmaxf(cm, C[(size_t)pi * HID + hd]);
        }
    }
    h0[k * HID + hd] = any ? hm : 0.0f;
    c0[k * HID + hd] = any ? cm : 0.0f;
}

// ---------------------------------------------------------------------------
// Root detection + final masked max/dot reduction
// ---------------------------------------------------------------------------
__global__ void k_count_children(const int* __restrict__ pidx,
                                 const unsigned char* __restrict__ pval,
                                 int* __restrict__ appears, int n)
{
    int i = blockIdx.x * blockDim.x + threadIdx.x;
    if (i < n && pval[i]) atomicAdd(&appears[pidx[i]], 1);
}

__global__ void k_final_reduce(const float* __restrict__ H,
                               const int* __restrict__ appears,
                               const float* __restrict__ Wlin,
                               const float* __restrict__ blin,
                               float* __restrict__ out)
{
    __shared__ float red[HID];
    const int hd = threadIdx.x;
    float m = -1e30f;
    for (int n = 0; n < NN; ++n)
        if (appears[n] == 0) m = fmaxf(m, H[(size_t)n * HID + hd]);
    red[hd] = m * Wlin[hd];
    __syncthreads();
    for (int s = HID / 2; s > 0; s >>= 1) {
        if (hd < s) red[hd] += red[hd + s];
        __syncthreads();
    }
    if (hd == 0) out[0] = red[0] + blin[0];
}

// ---------------------------------------------------------------------------
// host-side launch
// ---------------------------------------------------------------------------
extern "C" void kernel_launch(void* const* d_in, const int* in_sizes, int n_in,
                              void* d_out, int out_size, void* d_ws, size_t ws_size,
                              hipStream_t stream)
{
    (void)in_sizes; (void)n_in; (void)out_size; (void)ws_size;

    const float*         tokens  = (const float*)d_in[0];
    const int*           pidx    = (const int*)d_in[1];
    const unsigned char* pval    = (const unsigned char*)d_in[2];
    const float* Wih_tok = (const float*)d_in[3];
    const float* Whh_tok = (const float*)d_in[4];
    const float* bih_tok = (const float*)d_in[5];
    const float* bhh_tok = (const float*)d_in[6];
    const float* Wih_ins = (const float*)d_in[7];
    const float* Whh_ins = (const float*)d_in[8];
    const float* bih_ins = (const float*)d_in[9];
    const float* bhh_ins = (const float*)d_in[10];
    const float* Wlin    = (const float*)d_in[11];
    const float* blin    = (const float*)d_in[12];
    float* out = (float*)d_out;

    // ---- workspace carve-up (256B aligned) ----
    char* ws = (char*)d_ws;
    size_t off = 0;
    auto carve = [&](size_t bytes) -> char* {
        char* p = ws + off;
        off = (off + bytes + 255) & ~(size_t)255;
        return p;
    };
    _Float16* Wih_tok_p = (_Float16*)carve((size_t)G4 * EE  * 2);
    _Float16* Whh_tok_p = (_Float16*)carve((size_t)G4 * HID * 2);
    _Float16* Wih_ins_p = (_Float16*)carve((size_t)G4 * HID * 2);
    _Float16* Whh_ins_p = (_Float16*)carve((size_t)G4 * HID * 2);
    float* bsum_tok = (float*)carve((size_t)G4 * 4);
    float* bsum_ins = (float*)carve((size_t)G4 * 4);
    float* h_a    = (float*)carve((size_t)NN * HID * 4);   // h ping
    float* h_b    = (float*)carve((size_t)NN * HID * 4);   // h pong
    float* c_tok  = (float*)carve((size_t)NN * HID * 4);
    float* Hdag   = (float*)carve((size_t)NN * HID * 4);
    float* Cdag   = (float*)carve((size_t)NN * HID * 4);
    float* h0buf  = (float*)carve((size_t)KLVL * HID * 4);
    float* c0buf  = (float*)carve((size_t)KLVL * HID * 4);
    int*   appears= (int*)  carve((size_t)NN * 4);

    const int TPB = 256;
    auto blocks = [](int n) { return (n + 255) / 256; };

    // ---- 1. permute + convert weights/biases ----
    k_permute_w<<<blocks(G4 * EE),  TPB, 0, stream>>>(Wih_tok, Wih_tok_p, EE);
    k_permute_w<<<blocks(G4 * HID), TPB, 0, stream>>>(Whh_tok, Whh_tok_p, HID);
    k_permute_w<<<blocks(G4 * HID), TPB, 0, stream>>>(Wih_ins, Wih_ins_p, HID);
    k_permute_w<<<blocks(G4 * HID), TPB, 0, stream>>>(Whh_ins, Whh_ins_p, HID);
    k_permute_bias<<<blocks(G4), TPB, 0, stream>>>(bih_tok, bhh_tok, bsum_tok);
    k_permute_bias<<<blocks(G4), TPB, 0, stream>>>(bih_ins, bhh_ins, bsum_ins);

    // ---- 2. token LSTM (fused GEMM+cell, ping-pong h) ----
    k_zero_f32<<<blocks(NN * HID), TPB, 0, stream>>>(h_a,   NN * HID);
    k_zero_f32<<<blocks(NN * HID), TPB, 0, stream>>>(c_tok, NN * HID);
    float* h_in  = h_a;
    float* h_out = h_b;
    for (int t = 0; t < TT; ++t) {
        const float* xt = tokens + (size_t)t * EE;   // row n at xt + n*(TT*EE)
        k_gemm_lstm<<<dim3(NN / 64, G4 / 64), TPB, 0, stream>>>(
            xt, TT * EE, Wih_tok_p, EE,
            h_in, HID, Whh_tok_p, HID,
            bsum_tok, c_tok, h_out, c_tok);
        float* tmp = h_in; h_in = h_out; h_out = tmp;
    }
    float* ins_embed = h_in;   // result of the last step

    // ---- 3. DAG LSTM over levels (fused) ----
    for (int l = 0; l < LEVELS; ++l) {
        const int base = l * KLVL;
        k_gather_parents<<<KLVL, TPB, 0, stream>>>(
            pidx, pval, Hdag, Cdag, h0buf, c0buf, base);
        k_gemm_lstm<<<dim3(KLVL / 64, G4 / 64), TPB, 0, stream>>>(
            ins_embed + (size_t)base * HID, HID, Wih_ins_p, HID,
            h0buf, HID, Whh_ins_p, HID,
            bsum_ins, c0buf,
            Hdag + (size_t)base * HID, Cdag + (size_t)base * HID);
    }

    // ---- 4. roots + final reduction ----
    k_zero_i32<<<blocks(NN), TPB, 0, stream>>>(appears, NN);
    k_count_children<<<blocks(NN * PP), TPB, 0, stream>>>(pidx, pval, appears, NN * PP);
    k_final_reduce<<<1, HID, 0, stream>>>(Hdag, appears, Wlin, blin, out);
}
